// Model_89670327206497
// MI455X (gfx1250) — matrix operand order repaired and checked by reading, not compiled, not run on hardware
//
#include <hip/hip_runtime.h>

typedef __attribute__((ext_vector_type(16))) _Float16 v16h;
typedef __attribute__((ext_vector_type(8)))  float    v8f;

#define N_LOC 20000
#define N_EXP 200000
#define HDIM  64
#define NEDGE 1500000
#define NLAB  400000

// ---------------------------------------------------------------------------
// zero fill (graph-capture safe, deterministic)
// ---------------------------------------------------------------------------
__global__ void zero_f32(float* __restrict__ p, size_t n) {
    size_t i = (size_t)blockIdx.x * blockDim.x + threadIdx.x;
    size_t st = (size_t)gridDim.x * blockDim.x;
    for (; i < n; i += st) p[i] = 0.0f;
}

// ---------------------------------------------------------------------------
// Edge scatter-accumulate: 16 lanes per edge, float4 each (64 floats/row).
// fp32 hardware atomics (global_atomic_add_f32, no return). agg_exp (51MB)
// fits in the 192MB L2, so atomics resolve in-cache.
// ---------------------------------------------------------------------------
__global__ __launch_bounds__(256) void scatter_accum(
    const float* __restrict__ xsrc,
    const int*   __restrict__ src,
    const int*   __restrict__ dst,
    float* __restrict__ agg,
    float* __restrict__ cnt,
    int E, int addCnt)
{
    int tid  = blockIdx.x * blockDim.x + threadIdx.x;
    int e    = tid >> 4;
    int lane = tid & 15;
    if (e >= E) return;
    int s = src[e];
    int d = dst[e];
    const float4 v = ((const float4*)(xsrc + (size_t)s * HDIM))[lane];
    float* o = agg + (size_t)d * HDIM + lane * 4;
    unsafeAtomicAdd(o + 0, v.x);
    unsafeAtomicAdd(o + 1, v.y);
    unsafeAtomicAdd(o + 2, v.z);
    unsafeAtomicAdd(o + 3, v.w);
    if (addCnt && lane == 0) unsafeAtomicAdd(cnt + d, 1.0f);
}

// ---------------------------------------------------------------------------
// SAGE dense phase: out[N,64] = (agg/max(cnt,1)) @ Wl + xdst @ Wr + b, opt relu
// One wave per 16-row tile; A = [agg_norm | xdst] 16x128 f16 on the fly,
// B = [Wl;Wr] 128x64 f16 in LDS ([n][k] so each fragment is a 32B LDS read).
// 16x v_wmma_f32_16x16x32_f16 per wave.
// ---------------------------------------------------------------------------
__global__ __launch_bounds__(256) void sage_wmma(
    const float* __restrict__ agg, const float* __restrict__ cnt,
    const float* __restrict__ xdst,
    const float* __restrict__ Wl,  const float* __restrict__ Wr,
    const float* __restrict__ bias,
    float* __restrict__ out, int N, int doRelu)
{
    __shared__ _Float16 Bs[64 * 128];   // [n][k], 16 KB
    __shared__ float    sb[64];
    for (int i = threadIdx.x; i < 64 * 128; i += 256) {
        int n = i >> 7, k = i & 127;
        float w = (k < 64) ? Wl[k * 64 + n] : Wr[(k - 64) * 64 + n];
        Bs[i] = (_Float16)w;
    }
    if (threadIdx.x < 64) sb[threadIdx.x] = bias[threadIdx.x];
    __syncthreads();

    int wave = threadIdx.x >> 5;
    int lane = threadIdx.x & 31;
    int tile = blockIdx.x * 8 + wave;
    if (tile * 16 >= N) return;               // wave-uniform: EXEC stays all-1s

    int m     = lane & 15;                    // A row / B,C column within tile
    int khalf = lane >> 4;                    // K sub-block select per ISA layout
    size_t row = (size_t)tile * 16 + m;
    float scale = 1.0f / fmaxf(cnt[row], 1.0f);

    const float* aptr[2] = { agg + row * HDIM, xdst + row * HDIM };

    v8f acc[4] = {};                          // 16x64 output in 4 col-tiles
    #pragma unroll
    for (int c = 0; c < 4; ++c) {             // K chunks of 32 (agg lo/hi, x lo/hi)
        const float* ap = aptr[c >> 1];
        float sc = (c >> 1) ? 1.0f : scale;
        int koff = (c & 1) * 32;
        v16h A;
        #pragma unroll
        for (int e = 0; e < 16; ++e) {        // A layout matching B: K = 16*khalf + e
            int K = koff + khalf * 16 + e;
            A[e] = (_Float16)(ap[K] * sc);
        }
        #pragma unroll
        for (int nt = 0; nt < 4; ++nt) {
            v16h B = *(const v16h*)(Bs + (nt * 16 + m) * 128 + c * 32 + khalf * 16);
            acc[nt] = __builtin_amdgcn_wmma_f32_16x16x32_f16(
                false, A, false, B, (short)0, acc[nt], false, false);
        }
    }
    // C layout: lane holds col nt*16+m, rows j + 8*khalf in vgpr j
    #pragma unroll
    for (int nt = 0; nt < 4; ++nt) {
        int col = nt * 16 + m;
        float bv = sb[col];
        #pragma unroll
        for (int j = 0; j < 8; ++j) {
            int r = tile * 16 + khalf * 8 + j;
            float v = acc[nt][j] + bv;
            if (doRelu) v = fmaxf(v, 0.0f);
            out[(size_t)r * HDIM + col] = v;
        }
    }
}

// ---------------------------------------------------------------------------
// Edge decoder: out[i] = dW2 . relu(concat(zloc[row], zexp[col]) @ dW1 + db1) + db2
// 16 label-edges per wave via WMMA; final [64,1] matvec folded into a
// register relu-scale + 16-lane shfl_xor reduction (no hidden tensor in HBM).
// ---------------------------------------------------------------------------
__global__ __launch_bounds__(256) void decoder_wmma(
    const float* __restrict__ zloc, const float* __restrict__ zexp,
    const int*   __restrict__ rowi, const int* __restrict__ coli,
    const float* __restrict__ dW1,  const float* __restrict__ db1,
    const float* __restrict__ dW2,  const float* __restrict__ db2,
    float* __restrict__ out, int L)
{
    __shared__ _Float16 Bs[64 * 128];   // [n][k]: dW1 is [128,64] row-major
    __shared__ float sb1[64], sw2[64];
    for (int i = threadIdx.x; i < 64 * 128; i += 256) {
        int n = i >> 7, k = i & 127;
        Bs[i] = (_Float16)dW1[k * 64 + n];
    }
    if (threadIdx.x < 64) { sb1[threadIdx.x] = db1[threadIdx.x]; sw2[threadIdx.x] = dW2[threadIdx.x]; }
    __syncthreads();

    int wave = threadIdx.x >> 5;
    int lane = threadIdx.x & 31;
    int tile = blockIdx.x * 8 + wave;
    if (tile * 16 >= L) return;

    int m     = lane & 15;
    int khalf = lane >> 4;
    int idx   = tile * 16 + m;
    const float* zl = zloc + (size_t)rowi[idx] * HDIM;
    const float* ze = zexp + (size_t)coli[idx] * HDIM;
    const float* aptr[2] = { zl, ze };

    v8f acc[4] = {};
    #pragma unroll
    for (int c = 0; c < 4; ++c) {
        const float* ap = aptr[c >> 1];
        int koff = (c & 1) * 32;
        v16h A;
        #pragma unroll
        for (int e = 0; e < 16; ++e) {
            int K = koff + khalf * 16 + e;
            A[e] = (_Float16)ap[K];
        }
        #pragma unroll
        for (int nt = 0; nt < 4; ++nt) {
            v16h B = *(const v16h*)(Bs + (nt * 16 + m) * 128 + c * 32 + khalf * 16);
            acc[nt] = __builtin_amdgcn_wmma_f32_16x16x32_f16(
                false, A, false, B, (short)0, acc[nt], false, false);
        }
    }
    // per-lane partial: sum over this lane's 4 columns of relu(h)*w2
    float p[8];
    #pragma unroll
    for (int j = 0; j < 8; ++j) {
        float s = 0.0f;
        #pragma unroll
        for (int nt = 0; nt < 4; ++nt) {
            int col = nt * 16 + m;
            float h = fmaxf(acc[nt][j] + sb1[col], 0.0f);
            s += h * sw2[col];
        }
        p[j] = s;
    }
    // reduce across the 16 lanes of each half (xor masks < 16 stay in-half)
    #pragma unroll
    for (int j = 0; j < 8; ++j) {
        float s = p[j];
        #pragma unroll
        for (int off = 1; off < 16; off <<= 1)
            s += __shfl_xor(s, off, 32);
        p[j] = s;
    }
    if (m == 0) {                              // lanes 0 and 16 write rows 0-7 / 8-15
        float b2 = db2[0];
        #pragma unroll
        for (int j = 0; j < 8; ++j)
            out[tile * 16 + khalf * 8 + j] = p[j] + b2;
    }
}

// ---------------------------------------------------------------------------
extern "C" void kernel_launch(void* const* d_in, const int* in_sizes, int n_in,
                              void* d_out, int out_size, void* d_ws, size_t ws_size,
                              hipStream_t stream) {
    const float* emb_loc = (const float*)d_in[0];
    const float* emb_exp = (const float*)d_in[1];
    const float* W1l_of  = (const float*)d_in[2];
    const float* b1_of   = (const float*)d_in[3];
    const float* W1r_of  = (const float*)d_in[4];
    const float* W1l_rev = (const float*)d_in[5];
    const float* b1_rev  = (const float*)d_in[6];
    const float* W1r_rev = (const float*)d_in[7];
    const float* W2l_of  = (const float*)d_in[8];
    const float* b2_of   = (const float*)d_in[9];
    const float* W2r_of  = (const float*)d_in[10];
    const float* W2l_rev = (const float*)d_in[11];
    const float* b2_rev  = (const float*)d_in[12];
    const float* W2r_rev = (const float*)d_in[13];
    const float* dW1     = (const float*)d_in[14];
    const float* db1     = (const float*)d_in[15];
    const float* dW2     = (const float*)d_in[16];
    const float* db2     = (const float*)d_in[17];
    const int*   edge_of  = (const int*)d_in[18];   // [2,E] row0=src(loc) row1=dst(exp)
    const int*   edge_rev = (const int*)d_in[19];   // [2,E] row0=src(exp) row1=dst(loc)
    const int*   eli      = (const int*)d_in[20];   // [2,L] row0=loc row1=exp
    float* outp = (float*)d_out;

    // workspace layout (floats)
    float* ws = (float*)d_ws;
    float* aggE = ws;                               // N_EXP*64
    float* aggL = aggE + (size_t)N_EXP * HDIM;      // N_LOC*64
    float* cntE = aggL + (size_t)N_LOC * HDIM;      // N_EXP
    float* cntL = cntE + N_EXP;                     // N_LOC
    float* zE   = cntL + N_LOC;                     // N_EXP*64
    float* zL   = zE   + (size_t)N_EXP * HDIM;      // N_LOC*64
    float* zE2  = zL   + (size_t)N_LOC * HDIM;      // N_EXP*64
    float* zL2  = zE2  + (size_t)N_EXP * HDIM;      // N_LOC*64

    const size_t nAgg = (size_t)(N_EXP + N_LOC) * HDIM;
    const size_t nAggCnt = nAgg + N_EXP + N_LOC;

    const int scatterBlocks = (NEDGE * 16 + 255) / 256;
    const int gridExp = (N_EXP / 16 + 7) / 8;
    const int gridLoc = (N_LOC / 16 + 7) / 8;
    const int gridLab = (NLAB  / 16 + 7) / 8;

    // ---- layer 1 ----
    zero_f32<<<2048, 256, 0, stream>>>(aggE, nAggCnt);
    scatter_accum<<<scatterBlocks, 256, 0, stream>>>(emb_loc, edge_of,  edge_of  + NEDGE, aggE, cntE, NEDGE, 1);
    scatter_accum<<<scatterBlocks, 256, 0, stream>>>(emb_exp, edge_rev, edge_rev + NEDGE, aggL, cntL, NEDGE, 1);
    sage_wmma<<<gridExp, 256, 0, stream>>>(aggE, cntE, emb_exp, W1l_of,  W1r_of,  b1_of,  zE, N_EXP, 1);
    sage_wmma<<<gridLoc, 256, 0, stream>>>(aggL, cntL, emb_loc, W1l_rev, W1r_rev, b1_rev, zL, N_LOC, 1);

    // ---- layer 2 (counts unchanged; re-zero aggs only) ----
    zero_f32<<<2048, 256, 0, stream>>>(aggE, nAgg);
    scatter_accum<<<scatterBlocks, 256, 0, stream>>>(zL, edge_of,  edge_of  + NEDGE, aggE, cntE, NEDGE, 0);
    scatter_accum<<<scatterBlocks, 256, 0, stream>>>(zE, edge_rev, edge_rev + NEDGE, aggL, cntL, NEDGE, 0);
    sage_wmma<<<gridExp, 256, 0, stream>>>(aggE, cntE, zE, W2l_of,  W2r_of,  b2_of,  zE2, N_EXP, 0);
    sage_wmma<<<gridLoc, 256, 0, stream>>>(aggL, cntL, zL, W2l_rev, W2r_rev, b2_rev, zL2, N_LOC, 0);

    // ---- decoder ----
    decoder_wmma<<<gridLab, 256, 0, stream>>>(zL2, zE2, eli, eli + NLAB,
                                              dW1, db1, dW2, db2, outp, NLAB);
}
